// magnet_loss_60404420051124
// MI455X (gfx1250) — matrix operand order repaired and checked by reading, not compiled, not run on hardware
//
#include <hip/hip_runtime.h>
#include <hip/hip_bf16.h>

// ---------------------------------------------------------------------------
// Magnet loss, MI455X (gfx1250). Dominant GEMM uses v_wmma_f32_16x16x32_bf16
// with double-buffered LDS tiles filled by GLOBAL_LOAD_ASYNC_TO_LDS_B128
// (ASYNCcnt-tracked DMA, no VGPR round trip) when the builtins are available.
// Pipeline:
//   0) zero accumulators (counts, sums, lv_acc, scalars)
//   1) sample_pass : atomics segment-sum, counts, xnorm, X->bf16
//   2) means_pass  : means = sums/max(cnt,1), cnorm, means->bf16
//   3) gemm_d2     : WMMA bf16 128x128 tiles, fused d2 = max(xn+cn-2*dot, 0)
//   4) own_reduce  : sum_i d2[i, cid_i]      (variance input)
//   5) var_kernel  : variance, inv2v
//   6) row_pass    : sumE, denom/numer/term, scatter atomics
//   7) final_pass  : assemble (loss, loss_vector, loss_count, variance)
// d2 [N,K] = 64MB lives in workspace and fits in the 192MB L2.
// ---------------------------------------------------------------------------

typedef __attribute__((ext_vector_type(16))) __bf16 v16bf;
typedef __attribute__((ext_vector_type(8)))  float  v8f;
typedef __attribute__((ext_vector_type(4)))  int    i32x4;

#define ALPHA_ 7.18f
#define EPS_   1e-8f

#define BM 128
#define BN 128
#define BK 32
#define LDT 40   // padded LDS row length in bf16 elems (80B, 16B-aligned)

// 2 arrays * (BM*4/256 chunks per thread) = async issues per wave per tile
#define ISSUES_PER_TILE 4

#if defined(__has_builtin)
#if __has_builtin(__builtin_amdgcn_global_load_async_to_lds_b128) && \
    __has_builtin(__builtin_amdgcn_s_wait_asynccnt)
#define USE_ASYNC 1
#endif
#endif

// pointer-to-int4 in global (AS1) and LDS (AS3) address spaces, matching the
// builtin's parameter types reported by clang
typedef __attribute__((address_space(1))) i32x4* p1_v4i;
typedef __attribute__((address_space(3))) i32x4* p3_v4i;

__device__ __forceinline__ unsigned int f2bf(float f) {
  unsigned int u = __float_as_uint(f);
  return (u + 0x7FFFu + ((u >> 16) & 1u)) >> 16;   // round-to-nearest-even
}

// -------------------- 0: zero accumulators --------------------
__global__ void zero_f32(float* __restrict__ p, size_t n) {
  size_t i = (size_t)blockIdx.x * blockDim.x + threadIdx.x;
  if (i < n) p[i] = 0.0f;
}

// -------------------- 1: per-sample pass --------------------
__global__ __launch_bounds__(256)
void sample_pass(const float* __restrict__ X, const int* __restrict__ indices,
                 const int* __restrict__ assign, float* __restrict__ sums,
                 float* __restrict__ counts, float* __restrict__ xnorm,
                 unsigned short* __restrict__ Xb, int D) {
  __shared__ float red[256];
  const int i = blockIdx.x, t = threadIdx.x;
  const int cid = assign[indices[i]];
  const float* row = X + (size_t)i * D;
  float* srow = sums + (size_t)cid * D;
  unsigned short* brow = Xb + (size_t)i * D;
  float local = 0.0f;
  for (int d = t * 4; d < D; d += 256 * 4) {
    float4 v = *(const float4*)(row + d);
    atomicAdd(srow + d + 0, v.x);
    atomicAdd(srow + d + 1, v.y);
    atomicAdd(srow + d + 2, v.z);
    atomicAdd(srow + d + 3, v.w);
    uint2 p;
    p.x = f2bf(v.x) | (f2bf(v.y) << 16);
    p.y = f2bf(v.z) | (f2bf(v.w) << 16);
    *(uint2*)(brow + d) = p;
    local += v.x * v.x + v.y * v.y + v.z * v.z + v.w * v.w;
  }
  red[t] = local; __syncthreads();
  for (int s = 128; s > 0; s >>= 1) {
    if (t < s) red[t] += red[t + s];
    __syncthreads();
  }
  if (t == 0) { xnorm[i] = red[0]; atomicAdd(&counts[cid], 1.0f); }
}

// -------------------- 2: cluster means --------------------
__global__ __launch_bounds__(256)
void means_pass(const float* __restrict__ sums, const float* __restrict__ counts,
                unsigned short* __restrict__ Mb, float* __restrict__ cnorm, int D) {
  __shared__ float red[256];
  const int k = blockIdx.x, t = threadIdx.x;
  const float inv = 1.0f / fmaxf(counts[k], 1.0f);
  const float* srow = sums + (size_t)k * D;
  unsigned short* brow = Mb + (size_t)k * D;
  float local = 0.0f;
  for (int d = t * 4; d < D; d += 256 * 4) {
    float4 s = *(const float4*)(srow + d);
    float mx = s.x * inv, my = s.y * inv, mz = s.z * inv, mw = s.w * inv;
    uint2 p;
    p.x = f2bf(mx) | (f2bf(my) << 16);
    p.y = f2bf(mz) | (f2bf(mw) << 16);
    *(uint2*)(brow + d) = p;
    local += mx * mx + my * my + mz * mz + mw * mw;
  }
  red[t] = local; __syncthreads();
  for (int s = 128; s > 0; s >>= 1) {
    if (t < s) red[t] += red[t + s];
    __syncthreads();
  }
  if (t == 0) cnorm[k] = red[0];
}

// -------------------- 3: WMMA GEMM + fused d2 --------------------
__global__ __launch_bounds__(256)
void gemm_d2(const unsigned short* __restrict__ Xb,
             const unsigned short* __restrict__ Mb,
             const float* __restrict__ xnorm, const float* __restrict__ cnorm,
             float* __restrict__ d2, int K, int D) {
  __shared__ unsigned short As[2][BM * LDT];
  __shared__ unsigned short Bs[2][BN * LDT];

  const int tid  = threadIdx.x;
  const int lane = tid & 31;
  const int wave = tid >> 5;        // 0..7
  const int wm   = wave & 3;        // 4 waves along M (32 rows each)
  const int wn   = wave >> 2;       // 2 waves along N (64 cols each)
  const int tm   = blockIdx.x * BM; // sample-row base
  const int tn   = blockIdx.y * BN; // cluster-col base
  const int lr   = lane & 15;
  const int lh   = lane >> 4;

  v8f acc[2][4];
  const v8f vz = {0.f, 0.f, 0.f, 0.f, 0.f, 0.f, 0.f, 0.f};
#pragma unroll
  for (int a = 0; a < 2; ++a)
#pragma unroll
    for (int b = 0; b < 4; ++b) acc[a][b] = vz;

  // issue the async (or fallback sync) copies for one 128x32 A tile and one
  // 128x32 B tile into LDS buffer `buf`; 512 chunks of 16B per tile.
  auto issue_tile = [&](int buf, int kk) {
#pragma unroll
    for (int c = tid; c < BM * 4; c += 256) {
      const int r = c >> 2, cq = c & 3;
#ifdef USE_ASYNC
      __builtin_amdgcn_global_load_async_to_lds_b128(
          (p1_v4i)(Xb + (size_t)(tm + r) * D + kk + cq * 8),
          (p3_v4i)(&As[buf][r * LDT + cq * 8]), 0, 0);
      __builtin_amdgcn_global_load_async_to_lds_b128(
          (p1_v4i)(Mb + (size_t)(tn + r) * D + kk + cq * 8),
          (p3_v4i)(&Bs[buf][r * LDT + cq * 8]), 0, 0);
#else
      *(uint4*)(&As[buf][r * LDT + cq * 8]) =
          *(const uint4*)(Xb + (size_t)(tm + r) * D + kk + cq * 8);
      *(uint4*)(&Bs[buf][r * LDT + cq * 8]) =
          *(const uint4*)(Mb + (size_t)(tn + r) * D + kk + cq * 8);
#endif
    }
  };

  int buf = 0;
  issue_tile(0, 0);

  for (int kk = 0; kk < D; kk += BK) {
    const bool has_next = (kk + BK) < D;
    if (has_next) issue_tile(buf ^ 1, kk + BK);   // prefetch next tile

#ifdef USE_ASYNC
    // wait for current tile's copies; leave the prefetched tile in flight
    if (has_next) __builtin_amdgcn_s_wait_asynccnt(ISSUES_PER_TILE);
    else          __builtin_amdgcn_s_wait_asynccnt(0);
#endif
    __syncthreads();

    v16bf afrag[2], bfrag[4];
#pragma unroll
    for (int mi = 0; mi < 2; ++mi) {
      // A 16x32 bf16: M = lane%16, K-range = (lane/16)*16 .. +15 (contiguous)
      const int row = wm * 32 + mi * 16 + lr;
      const int kb  = lh * 16;
      union { uint4 u[2]; v16bf v; } tA;
      tA.u[0] = *(const uint4*)(&As[buf][row * LDT + kb]);
      tA.u[1] = *(const uint4*)(&As[buf][row * LDT + kb + 8]);
      afrag[mi] = tA.v;
    }
#pragma unroll
    for (int nj = 0; nj < 4; ++nj) {
      // B 32x16 bf16: N = lane%16, K-range = (lane/16)*16 .. +15 (contiguous)
      const int col = wn * 64 + nj * 16 + lr;
      const int kb  = lh * 16;
      union { uint4 u[2]; v16bf v; } tB;
      tB.u[0] = *(const uint4*)(&Bs[buf][col * LDT + kb]);
      tB.u[1] = *(const uint4*)(&Bs[buf][col * LDT + kb + 8]);
      bfrag[nj] = tB.v;
    }

#pragma unroll
    for (int mi = 0; mi < 2; ++mi)
#pragma unroll
      for (int nj = 0; nj < 4; ++nj)
        acc[mi][nj] = __builtin_amdgcn_wmma_f32_16x16x32_bf16(
            false, afrag[mi], false, bfrag[nj], (short)0, acc[mi][nj],
            false, false);

    __syncthreads();   // all waves done reading buf before it is refilled
    buf ^= 1;
  }

  // fused epilogue: d2 = max(||x||^2 + ||c||^2 - 2*dot, 0)
#pragma unroll
  for (int mi = 0; mi < 2; ++mi) {
#pragma unroll
    for (int nj = 0; nj < 4; ++nj) {
      const int col = tn + wn * 64 + nj * 16 + lr;
      const float cn = cnorm[col];
#pragma unroll
      for (int r = 0; r < 8; ++r) {
        // C/D layout: VGPR r holds M=r (lanes 0-15) / M=8+r (lanes 16-31)
        const int grow = tm + wm * 32 + mi * 16 + lh * 8 + r;
        const float v = xnorm[grow] + cn - 2.0f * acc[mi][nj][r];
        d2[(size_t)grow * K + col] = fmaxf(v, 0.0f);
      }
    }
  }
}

// -------------------- 4: sum of own distances --------------------
__global__ __launch_bounds__(256)
void own_reduce(const float* __restrict__ d2, const int* __restrict__ indices,
                const int* __restrict__ assign, float* __restrict__ sum_own, int K) {
  __shared__ float red[256];
  const int i = blockIdx.x * 256 + threadIdx.x;
  const int cid = assign[indices[i]];
  red[threadIdx.x] = d2[(size_t)i * K + cid];
  __syncthreads();
  for (int s = 128; s > 0; s >>= 1) {
    if (threadIdx.x < s) red[threadIdx.x] += red[threadIdx.x + s];
    __syncthreads();
  }
  if (threadIdx.x == 0) atomicAdd(sum_own, red[0]);
}

// -------------------- 5: variance scalars --------------------
__global__ void var_kernel(float* __restrict__ sc, int N) {
  const float stdev = sc[0] / (float)(N - 1);
  const float variance = stdev * stdev;
  sc[2] = variance;
  sc[3] = 1.0f / (2.0f * variance + EPS_);
}

// -------------------- 6: exp row pass --------------------
__global__ __launch_bounds__(256)
void row_pass(const float* __restrict__ d2, const int* __restrict__ indices,
              const int* __restrict__ assign, const float* __restrict__ sc,
              float* __restrict__ sum_term, float* __restrict__ lv_acc, int K) {
  __shared__ float red[256];
  const int i = blockIdx.x, t = threadIdx.x;
  const int cid = assign[indices[i]];
  const float inv2v = sc[3];
  const float* row = d2 + (size_t)i * K;
  float local = 0.0f;
  for (int k = t; k < K; k += 256) local += expf(-row[k] * inv2v);
  red[t] = local; __syncthreads();
  for (int s = 128; s > 0; s >>= 1) {
    if (t < s) red[t] += red[t + s];
    __syncthreads();
  }
  if (t == 0) {
    const float own   = row[cid];
    const float own_e = expf(-own * inv2v);
    const float denom = red[0] - own_e + EPS_;
    const float numer = expf(-own * inv2v - ALPHA_);
    const float term  = fmaxf(logf(numer / denom + EPS_), 0.0f);
    atomicAdd(sum_term, term);
    atomicAdd(&lv_acc[cid], -term);
  }
}

// -------------------- 7: finalize --------------------
__global__ void final_pass(float* __restrict__ out, const float* __restrict__ in_lv,
                           const float* __restrict__ in_lc,
                           const float* __restrict__ lv_acc,
                           const float* __restrict__ counts,
                           const float* __restrict__ sc, int N, int K) {
  const int i = blockIdx.x * blockDim.x + threadIdx.x;
  const float invN = 1.0f / (float)N;
  if (i == 0) {
    out[0] = -sc[1] * invN;      // loss
    out[1 + 2 * K] = sc[2];      // variance
  }
  if (i < K) {
    out[1 + i]     = (in_lv[i] + lv_acc[i]) * invN;  // loss_vector
    out[1 + K + i] = in_lc[i] + counts[i];           // loss_count
  }
}

// ---------------------------------------------------------------------------
extern "C" void kernel_launch(void* const* d_in, const int* in_sizes, int n_in,
                              void* d_out, int out_size, void* d_ws, size_t ws_size,
                              hipStream_t stream) {
  const float* X        = (const float*)d_in[0];  // [N*D]
  const int*   indices  = (const int*)d_in[1];    // [N]
  const int*   assign   = (const int*)d_in[2];    // [N]
  const float* in_lv    = (const float*)d_in[3];  // [K]
  const float* in_lc    = (const float*)d_in[4];  // [K]
  float* out = (float*)d_out;

  const int N = in_sizes[1];
  const int D = in_sizes[0] / N;
  const int K = in_sizes[3];

  // workspace layout (256B-aligned blocks); zero-region first
  size_t off = 0;
  auto take = [&](size_t bytes) -> size_t {
    size_t o = off; off = (off + bytes + 255) & ~(size_t)255; return o;
  };
  const size_t o_sc   = take(64 * sizeof(float));                 // scalars
  const size_t o_cnt  = take((size_t)K * sizeof(float));          // counts
  const size_t o_lv   = take((size_t)K * sizeof(float));          // lv_acc
  const size_t o_sums = take((size_t)K * D * sizeof(float));      // sums
  const size_t zero_bytes = off;
  const size_t o_cn   = take((size_t)K * sizeof(float));          // cnorm
  const size_t o_xn   = take((size_t)N * sizeof(float));          // xnorm
  const size_t o_Mb   = take((size_t)K * D * sizeof(unsigned short));
  const size_t o_Xb   = take((size_t)N * D * sizeof(unsigned short));
  const size_t o_d2   = take((size_t)N * (size_t)K * sizeof(float));
  (void)o_d2; (void)ws_size; (void)n_in; (void)out_size;

  char* w = (char*)d_ws;
  float*          sc     = (float*)(w + o_sc);
  float*          counts = (float*)(w + o_cnt);
  float*          lv_acc = (float*)(w + o_lv);
  float*          sums   = (float*)(w + o_sums);
  float*          cnorm  = (float*)(w + o_cn);
  float*          xnorm  = (float*)(w + o_xn);
  unsigned short* Mb     = (unsigned short*)(w + o_Mb);
  unsigned short* Xb     = (unsigned short*)(w + o_Xb);
  float*          d2     = (float*)(w + o_d2);

  const size_t zero_n = zero_bytes / sizeof(float);
  zero_f32<<<dim3((unsigned)((zero_n + 255) / 256)), 256, 0, stream>>>(sc, zero_n);

  sample_pass<<<dim3(N), 256, 0, stream>>>(X, indices, assign, sums, counts,
                                           xnorm, Xb, D);
  means_pass<<<dim3(K), 256, 0, stream>>>(sums, counts, Mb, cnorm, D);

  gemm_d2<<<dim3(N / BM, K / BN), 256, 0, stream>>>(Xb, Mb, xnorm, cnorm, d2, K, D);

  own_reduce<<<dim3(N / 256), 256, 0, stream>>>(d2, indices, assign, &sc[0], K);
  var_kernel<<<1, 1, 0, stream>>>(sc, N);
  row_pass<<<dim3(N), 256, 0, stream>>>(d2, indices, assign, sc, &sc[1], lv_acc, K);
  final_pass<<<dim3((2 * K + 2 + 255) / 256), 256, 0, stream>>>(
      out, in_lv, in_lc, lv_acc, counts, sc, N, K);
}
